// DeepBlockDense_15393162788947
// MI455X (gfx1250) — matrix operand order repaired and checked by reading, not compiled, not run on hardware
//
#include <hip/hip_runtime.h>

// Grouped 3-layer MLP: per group g: relu(relu(relu(x Wg1) Wg2) Wg3)
// - split-precision bf16 WMMA (hi/lo decomposition, 3 products -> ~fp32 accuracy)
// - one workgroup per group; all weights staged split+transposed in LDS (128 KB)
// - TDM (tensor_load_to_lds, 6-arg clang-23 form) double-buffers the fp32 x tile
// - dual accumulators per wave reuse A fragments across two output columns

typedef __attribute__((ext_vector_type(16))) __bf16       v16bf;
typedef __attribute__((ext_vector_type(8)))  float        v8f;
typedef __attribute__((ext_vector_type(4)))  unsigned int v4u;
typedef __attribute__((ext_vector_type(8)))  int          v8i;
typedef __attribute__((ext_vector_type(4)))  int          v4i;

#define G_   512
#define W_   64
#define H1_  128
#define H2_  128
#define H3_  64
#define B_   1024
#define BT   64          // batch rows per tile iteration
#define NTHREADS 256
#define NWAVES   8
#define NITER (B_ / BT)

union Frag { v16bf v; uint4 q[2]; };

__device__ __forceinline__ unsigned short f32_to_bf16(float f) {
  unsigned u = __float_as_uint(f);
  u += 0x7FFFu + ((u >> 16) & 1u);          // round-to-nearest-even
  return (unsigned short)(u >> 16);
}
__device__ __forceinline__ float bf16_to_f32(unsigned short h) {
  return __uint_as_float(((unsigned)h) << 16);
}
__device__ __forceinline__ void split2(float f, unsigned short& hi, unsigned short& lo) {
  hi = f32_to_bf16(f);
  lo = f32_to_bf16(f - bf16_to_f32(hi));
}

// A-operand (16-bit 16x32): lane holds row m0 + lane%16;
// VGPR0-3 = K kb..kb+7, VGPR4-7 = K kb+16..kb+23, kb = 8*(lane/16).
__device__ __forceinline__ v16bf load_fragA(const unsigned short* base, int ldk,
                                            int m0, int kc, int lane) {
  int m  = m0 + (lane & 15);
  int kb = kc + ((lane & 16) >> 1);
  const unsigned short* p = base + m * ldk + kb;
  Frag f;
  f.q[0] = *(const uint4*)(p);
  f.q[1] = *(const uint4*)(p + 16);
  return f.v;
}

// B-operand (16-bit 32x16): lane holds col n0 + lane%16;
// VGPR0-3 = K kb..kb+7, VGPR4-7 = K kb+16..kb+23, kb = 8*(lane/16); read from transposed (K-contiguous) copy.
__device__ __forceinline__ v16bf load_fragB(const unsigned short* baseT, int ldk,
                                            int n0, int kc, int lane) {
  int n  = n0 + (lane & 15);
  int kb = kc + ((lane & 16) >> 1);
  const unsigned short* p = baseT + n * ldk + kb;
  Frag f;
  f.q[0] = *(const uint4*)(p);
  f.q[1] = *(const uint4*)(p + 16);
  return f.v;
}

__device__ __forceinline__ v8f wmma_bf16(v16bf a, v16bf b, v8f c) {
  return __builtin_amdgcn_wmma_f32_16x16x32_bf16(false, a, false, b,
                                                 (short)0, c, false, false);
}

// TDM: 2D tile (BT rows x W_ fp32 elements, row stride G_*W_ elements) -> LDS.
// D# per CDNA5 ISA 8.3/8.4: group0 = {count/type, lds_addr, global_addr},
// group1 = {data_size, tensor dims, tile dims, dim0 stride}.
__device__ __forceinline__ void tdm_load_x_tile(const float* gsrc, unsigned lds_byte) {
  unsigned long long ga = (unsigned long long)(uintptr_t)gsrc;
  v4u g0;
  g0[0] = 1u;                                           // count=1 (valid descriptor)
  g0[1] = lds_byte;                                     // lds_addr [63:32]
  g0[2] = (unsigned)ga;                                 // global_addr [95:64]
  g0[3] = (unsigned)((ga >> 32) & 0x01FFFFFFu) | (2u << 30);  // addr[56:32] | type=2
  v8i g1;
  g1[0] = (int)(2u << 16);                              // data_size = 4 bytes
  g1[1] = (int)((unsigned)W_ << 16);                    // tensor_dim0[15:0] @ [63:48]
  g1[2] = (int)((unsigned)BT << 16);                    // tensor_dim1[15:0] @ [95:80]
  g1[3] = (int)((unsigned)W_ << 16);                    // tile_dim0 @ [127:112]
  g1[4] = (int)BT;                                      // tile_dim1 @ [143:128]
  g1[5] = (int)(G_ * W_);                               // tensor_dim0_stride[31:0]
  g1[6] = 0;
  g1[7] = 0;
  v4i z4 = {0, 0, 0, 0};
  v8i z8 = {0, 0, 0, 0, 0, 0, 0, 0};
  __builtin_amdgcn_tensor_load_to_lds(g0, g1, z4, z4, z8, 0);  // 6-arg (clang-23)
}

// One GEMM layer: C = relu(A x B), split accumulation, dual accumulators per wave.
// Wave w: row stripe (w&3)*16, column tiles [(w>>2)*(NT/2), ...) taken in pairs.
template<int MT, int NT, int KK, bool TO_GLOBAL>
__device__ __forceinline__ void gemm_layer(
    const unsigned short* Ah, const unsigned short* Al,
    const unsigned short* Bh, const unsigned short* Bl,
    unsigned short* Ch, unsigned short* Cl,
    float* __restrict__ outp, int row0, int colbase,
    int wave, int lane) {
  static_assert(MT == 4 && (NT % 2) == 0, "tiling assumes 8 waves, MT=4");
  const int rm = (wave & 3) * 16;
  const int cb = (wave >> 2) * (NT / 2);
  const int rbase = rm + ((lane & 16) >> 1);
  const int cl15  = (lane & 15);
  for (int j = 0; j < NT / 2; j += 2) {
    const int tn0 = (cb + j) * 16;
    const int tn1 = tn0 + 16;
    v8f acc0 = {}, acc1 = {};
#pragma unroll
    for (int kc = 0; kc < KK; kc += 32) {
      v16bf ah  = load_fragA(Ah, KK, rm, kc, lane);
      v16bf al  = load_fragA(Al, KK, rm, kc, lane);
      v16bf bh0 = load_fragB(Bh, KK, tn0, kc, lane);
      v16bf bl0 = load_fragB(Bl, KK, tn0, kc, lane);
      v16bf bh1 = load_fragB(Bh, KK, tn1, kc, lane);
      v16bf bl1 = load_fragB(Bl, KK, tn1, kc, lane);
      acc0 = wmma_bf16(ah, bh0, acc0);   // interleaved: independent chains
      acc1 = wmma_bf16(ah, bh1, acc1);
      acc0 = wmma_bf16(al, bh0, acc0);
      acc1 = wmma_bf16(al, bh1, acc1);
      acc0 = wmma_bf16(ah, bl0, acc0);
      acc1 = wmma_bf16(ah, bl1, acc1);
    }
    if (TO_GLOBAL) {
#pragma unroll
      for (int v = 0; v < 8; ++v) {
        float r0 = acc0[v] > 0.f ? acc0[v] : 0.f;
        float r1 = acc1[v] > 0.f ? acc1[v] : 0.f;
        size_t ro = (size_t)(row0 + rbase + v) * (G_ * H3_) + colbase;
        outp[ro + tn0 + cl15] = r0;
        outp[ro + tn1 + cl15] = r1;
      }
    } else {
#pragma unroll
      for (int v = 0; v < 8; ++v) {
        float r0 = acc0[v] > 0.f ? acc0[v] : 0.f;
        float r1 = acc1[v] > 0.f ? acc1[v] : 0.f;
        unsigned short h, l;
        int ro = (rbase + v) * (NT * 16);
        split2(r0, h, l); Ch[ro + tn0 + cl15] = h; Cl[ro + tn0 + cl15] = l;
        split2(r1, h, l); Ch[ro + tn1 + cl15] = h; Cl[ro + tn1 + cl15] = l;
      }
    }
  }
}

// fp32 [K][N] global -> split bf16 [N][K] LDS (transposed: B K-runs contiguous)
__device__ __forceinline__ void stage_weightT(const float* __restrict__ src,
                                              unsigned short* dH, unsigned short* dL,
                                              int K, int N, int tid) {
  for (int i = tid * 4; i < K * N; i += NTHREADS * 4) {
    float4 w = *(const float4*)&src[i];     // coalesced b128 reads
    int k = i / N, n = i % N;
    unsigned short hi, lo;
    split2(w.x, hi, lo); dH[(n + 0) * K + k] = hi; dL[(n + 0) * K + k] = lo;
    split2(w.y, hi, lo); dH[(n + 1) * K + k] = hi; dL[(n + 1) * K + k] = lo;
    split2(w.z, hi, lo); dH[(n + 2) * K + k] = hi; dL[(n + 2) * K + k] = lo;
    split2(w.w, hi, lo); dH[(n + 3) * K + k] = hi; dL[(n + 3) * K + k] = lo;
  }
}

__global__ __launch_bounds__(NTHREADS)
void grouped_mlp3_kernel(const float* __restrict__ x,
                         const float* __restrict__ W1,
                         const float* __restrict__ W2,
                         const float* __restrict__ W3,
                         float* __restrict__ out) {
  __shared__ __align__(16) unsigned short sW1h[H1_ * W_],  sW1l[H1_ * W_];   // 32 KB
  __shared__ __align__(16) unsigned short sW2h[H2_ * H1_], sW2l[H2_ * H1_];  // 64 KB
  __shared__ __align__(16) unsigned short sW3h[H3_ * H2_], sW3l[H3_ * H2_];  // 32 KB
  __shared__ __align__(16) unsigned short sXh [BT * W_],   sXl [BT * W_];    // 16 KB
  __shared__ __align__(16) unsigned short sH1h[BT * H1_],  sH1l[BT * H1_];   // 32 KB
  __shared__ __align__(16) unsigned short sH2h[BT * H2_],  sH2l[BT * H2_];   // 32 KB
  __shared__ __align__(16) float sXraw[2][BT * W_];                          // 32 KB (TDM dbuf)

  const int g    = blockIdx.x;
  const int tid  = threadIdx.x;
  const int wave = tid >> 5;
  const int lane = tid & 31;
  const float* xg = x + (size_t)g * W_;

  // Prologue: kick off TDM for tile 0, then stage weights while the DMA runs.
  if (wave == 0)
    tdm_load_x_tile(xg, (unsigned)(uintptr_t)&sXraw[0][0]);

  stage_weightT(W1 + (size_t)g * W_  * H1_, sW1h, sW1l, W_,  H1_, tid);
  stage_weightT(W2 + (size_t)g * H1_ * H2_, sW2h, sW2l, H1_, H2_, tid);
  stage_weightT(W3 + (size_t)g * H2_ * H3_, sW3h, sW3l, H2_, H3_, tid);
  __syncthreads();

  for (int it = 0; it < NITER; ++it) {
    const int b0  = it * BT;
    const int buf = it & 1;

    if (wave == 0) __builtin_amdgcn_s_wait_tensorcnt((short)0);  // tile `it` landed
    __syncthreads();

    // Prefetch next tile into the other buffer while we compute this one.
    if (wave == 0 && it + 1 < NITER)
      tdm_load_x_tile(xg + (size_t)(b0 + BT) * (G_ * W_),
                      (unsigned)(uintptr_t)&sXraw[buf ^ 1][0]);

    // Convert staged fp32 tile -> split bf16 planes (vectorized).
    for (int i = tid * 4; i < BT * W_; i += NTHREADS * 4) {
      float4 xv = *(const float4*)&sXraw[buf][i];
      unsigned short h0, l0, h1, l1, h2, l2, h3, l3;
      split2(xv.x, h0, l0); split2(xv.y, h1, l1);
      split2(xv.z, h2, l2); split2(xv.w, h3, l3);
      uint2 hv, lv;
      hv.x = (unsigned)h0 | ((unsigned)h1 << 16);
      hv.y = (unsigned)h2 | ((unsigned)h3 << 16);
      lv.x = (unsigned)l0 | ((unsigned)l1 << 16);
      lv.y = (unsigned)l2 | ((unsigned)l3 << 16);
      *(uint2*)&sXh[i] = hv;
      *(uint2*)&sXl[i] = lv;
    }
    __syncthreads();

    gemm_layer<BT/16, H1_/16, W_,  false>(sXh,  sXl,  sW1h, sW1l, sH1h, sH1l,
                                          nullptr, 0, 0, wave, lane);
    __syncthreads();
    gemm_layer<BT/16, H2_/16, H1_, false>(sH1h, sH1l, sW2h, sW2l, sH2h, sH2l,
                                          nullptr, 0, 0, wave, lane);
    __syncthreads();
    gemm_layer<BT/16, H3_/16, H2_, true >(sH2h, sH2l, sW3h, sW3l, nullptr, nullptr,
                                          out, b0, g * H3_, wave, lane);
    __syncthreads();   // protect sX / sXraw rewrite next iteration
  }
}

extern "C" void kernel_launch(void* const* d_in, const int* in_sizes, int n_in,
                              void* d_out, int out_size, void* d_ws, size_t ws_size,
                              hipStream_t stream) {
  (void)in_sizes; (void)n_in; (void)out_size; (void)d_ws; (void)ws_size;
  const float* x  = (const float*)d_in[0];
  const float* W1 = (const float*)d_in[1];
  const float* W2 = (const float*)d_in[2];
  const float* W3 = (const float*)d_in[3];
  float* out = (float*)d_out;
  grouped_mlp3_kernel<<<dim3(G_), dim3(NTHREADS), 0, stream>>>(x, W1, W2, W3, out);
}
